// PAM_43602507989302
// MI455X (gfx1250) — matrix operand order repaired and checked by reading, not compiled, not run on hardware
//
#include <hip/hip_runtime.h>
#include <hip/hip_bf16.h>

// ---------------------------------------------------------------------------
// PAM position-attention, fused bf16-WMMA flash-attention for gfx1250 (MI455X)
// B=4, H=W=64 (N=4096), C=512, CR=64
// ---------------------------------------------------------------------------

typedef __attribute__((ext_vector_type(16))) __bf16 v16bf;
typedef __attribute__((ext_vector_type(8)))  __bf16 v8bf;
typedef __attribute__((ext_vector_type(8)))  float  v8f;
typedef __bf16 bf16_t;

#define WMMA_BF16(a, b, c) \
    __builtin_amdgcn_wmma_f32_16x16x32_bf16(false, (a), false, (b), (short)0, (c), false, false)

static constexpr int kB  = 4;
static constexpr int kN  = 4096;   // H*W
static constexpr int kC  = 512;
static constexpr int kCR = 64;
static constexpr int kM  = kB * kN;        // 16384 total pixels
static constexpr int kNCOL = kCR + kCR + kC; // 640 concatenated proj columns

__device__ __forceinline__ v8f zero8() {
    v8f z;
#pragma unroll
    for (int i = 0; i < 8; ++i) z[i] = 0.0f;
    return z;
}

// ---------------------------------------------------------------------------
// Kernel 0: Wt[col][k] (bf16, k-contiguous) from Wb|Wc|Wd (f32, [k][col])
// ---------------------------------------------------------------------------
__global__ void pam_prep_weights(const float* __restrict__ Wb,
                                 const float* __restrict__ Wc,
                                 const float* __restrict__ Wd,
                                 bf16_t* __restrict__ Wt) {
    int i = blockIdx.x * blockDim.x + threadIdx.x;
    if (i >= kNCOL * kC) return;
    int col = i / kC;
    int k   = i % kC;
    float v;
    if (col < kCR)            v = Wb[k * kCR + col];
    else if (col < 2 * kCR)   v = Wc[k * kCR + (col - kCR)];
    else                      v = Wd[k * kC  + (col - 2 * kCR)];
    Wt[col * kC + k] = (bf16_t)v;
}

// ---------------------------------------------------------------------------
// Kernel 1: projections.  Each wave: one 16x64 output tile of [X * Wcat].
//   cols   0..63  -> Qb [BN][64] bf16 (row major)
//   cols  64..127 -> Kb [BN][64] bf16 (row major)
//   cols 128..639 -> Vt [B][C][N] bf16 (TRANSPOSED: contiguous along pixels)
// ---------------------------------------------------------------------------
__global__ void __launch_bounds__(128)
pam_proj(const float* __restrict__ x, const bf16_t* __restrict__ Wt,
         bf16_t* __restrict__ Qb, bf16_t* __restrict__ Kb, bf16_t* __restrict__ Vt) {
    const int lane = threadIdx.x & 31;
    const int wv   = threadIdx.x >> 5;
    const int tile = blockIdx.x * 4 + wv;        // 1024 mtiles x 10 ntiles
    const int nt   = tile % 10;
    const int mt   = tile / 10;
    const int m0   = mt * 16;                    // pixel tile base
    const int n0   = nt * 64;                    // column tile base (0..639)
    const int lrow = lane & 15;
    const int hsel = (lane < 16) ? 0 : 16;       // A-frag K split
    const int bsel = (lane < 16) ? 0 : 16;       // B-frag K split

    v8f acc[4];
#pragma unroll
    for (int i = 0; i < 4; ++i) acc[i] = zero8();

    const float* xrow = x + (size_t)(m0 + lrow) * kC;

    for (int kt = 0; kt < kC; kt += 32) {
        // A fragment: 16 rows x 32 k, f32 -> bf16 (RNE via __bf16 cast)
        v16bf a;
        const float* xa = xrow + kt + hsel;
#pragma unroll
        for (int i = 0; i < 8; ++i) {
            a[i]     = (bf16_t)xa[i];            // K = kt + hsel + i
            a[i + 8] = (bf16_t)xa[8 + i];        // K = kt + hsel + 8 + i
        }
#pragma unroll
        for (int cg = 0; cg < 4; ++cg) {
            const int col = n0 + cg * 16 + lrow;
            v16bf bfrag = *(const v16bf*)(Wt + (size_t)col * kC + kt + bsel);
            acc[cg] = WMMA_BF16(a, bfrag, acc[cg]);
        }
    }

    const int rb = (lane < 16) ? 0 : 8;          // D-frag row base
    if (n0 < 2 * kCR) {
        bf16_t* dst = (n0 < kCR) ? Qb : Kb;
        const int dbase = (n0 < kCR) ? n0 : (n0 - kCR);
#pragma unroll
        for (int cg = 0; cg < 4; ++cg) {
            const int dcol = dbase + cg * 16 + lrow;
#pragma unroll
            for (int e = 0; e < 8; ++e)
                dst[(size_t)(m0 + rb + e) * kCR + dcol] = (bf16_t)acc[cg][e];
        }
    } else {
        const int b    = m0 >> 12;               // pixel tiles never straddle batches
        const int nloc = (m0 & (kN - 1)) + rb;
#pragma unroll
        for (int cg = 0; cg < 4; ++cg) {
            const int c = (n0 - 2 * kCR) + cg * 16 + lrow;
            v8bf v;
#pragma unroll
            for (int e = 0; e < 8; ++e) v[e] = (bf16_t)acc[cg][e];
            *(v8bf*)(Vt + ((size_t)(b * kC + c) * kN) + nloc) = v;  // 16B contiguous
        }
    }
}

// ---------------------------------------------------------------------------
// Kernel 2: fused flash attention.
// Workgroup = one 16-query tile of one batch; 4 waves, each wave owns 128
// output channels. Online softmax over 32-key tiles. P goes D->A layout
// through a private 1KB LDS buffer per wave.
// ---------------------------------------------------------------------------
__global__ void __launch_bounds__(128)
pam_attn(const float* __restrict__ x, const float* __restrict__ gamma_p,
         const bf16_t* __restrict__ Qb, const bf16_t* __restrict__ Kb,
         const bf16_t* __restrict__ Vt, float* __restrict__ out) {
    const int lane = threadIdx.x & 31;
    const int wv   = threadIdx.x >> 5;
    const int b    = blockIdx.x >> 8;            // 256 query tiles per batch
    const int q0   = (blockIdx.x & 255) << 4;
    const int p0   = (b << 12) + q0;             // global query pixel base
    const int lrow = lane & 15;
    const int hsel = (lane < 16) ? 0 : 16;
    const int bsel = (lane < 16) ? 0 : 16;
    const int rb   = (lane < 16) ? 0 : 8;
    const int cb   = wv * 128;                   // this wave's channel base

    __shared__ bf16_t Ptile[4][16 * 32];
    bf16_t* myP = &Ptile[wv][0];

    // Q A-fragments: d in [0,32) and [32,64)
    v16bf qa[2];
    {
        const bf16_t* qr = Qb + (size_t)(p0 + lrow) * kCR;
#pragma unroll
        for (int f = 0; f < 2; ++f) {
            const bf16_t* pb = qr + 32 * f + hsel;
            v8bf lo = *(const v8bf*)(pb);
            v8bf hi = *(const v8bf*)(pb + 8);
#pragma unroll
            for (int i = 0; i < 8; ++i) { qa[f][i] = lo[i]; qa[f][i + 8] = hi[i]; }
        }
    }

    v8f o[8];
#pragma unroll
    for (int i = 0; i < 8; ++i) o[i] = zero8();
    float mrow[8], lsum[8];
#pragma unroll
    for (int e = 0; e < 8; ++e) { mrow[e] = -3.0e38f; lsum[e] = 0.0f; }

    const bf16_t* Kbb = Kb + (size_t)(b << 12) * kCR;
    const bf16_t* Vtb = Vt + (size_t)b * kC * kN;

    for (int kt = 0; kt < kN; kt += 32) {
        // ---- S = Q * K^T : two 16x16 frags over key halves, K-dim = 64 ----
        v8f s[2];
#pragma unroll
        for (int h = 0; h < 2; ++h) {
            const bf16_t* kr = Kbb + (size_t)(kt + h * 16 + lrow) * kCR;
            v16bf b0 = *(const v16bf*)(kr + bsel);        // d 0..31
            v16bf b1 = *(const v16bf*)(kr + 32 + bsel);   // d 32..63
            v8f z = zero8();
            z    = WMMA_BF16(qa[0], b0, z);
            s[h] = WMMA_BF16(qa[1], b1, z);
        }
        // ---- online softmax (rows live in one 16-lane half; xor<16 stays) --
#pragma unroll
        for (int e = 0; e < 8; ++e) {
            float t = fmaxf(s[0][e], s[1][e]);
            t = fmaxf(t, __shfl_xor(t, 1));
            t = fmaxf(t, __shfl_xor(t, 2));
            t = fmaxf(t, __shfl_xor(t, 4));
            t = fmaxf(t, __shfl_xor(t, 8));
            const float mn = fmaxf(mrow[e], t);
            const float sc = __expf(mrow[e] - mn);
            mrow[e] = mn;
            const float e0 = __expf(s[0][e] - mn);
            const float e1 = __expf(s[1][e] - mn);
            s[0][e] = e0; s[1][e] = e1;
            float rs = e0 + e1;
            rs += __shfl_xor(rs, 1);
            rs += __shfl_xor(rs, 2);
            rs += __shfl_xor(rs, 4);
            rs += __shfl_xor(rs, 8);
            lsum[e] = lsum[e] * sc + rs;
#pragma unroll
            for (int cg = 0; cg < 8; ++cg) o[cg][e] *= sc;  // rescale row e
        }
        // ---- P: D-layout -> A-layout via per-wave LDS ----------------------
#pragma unroll
        for (int e = 0; e < 8; ++e) {
            myP[(rb + e) * 32 + lrow]      = (bf16_t)s[0][e];
            myP[(rb + e) * 32 + 16 + lrow] = (bf16_t)s[1][e];
        }
        asm volatile("s_wait_dscnt 0" ::: "memory");
        v16bf pa;
        {
            const bf16_t* pr = myP + lrow * 32 + hsel;
            v8bf lo = *(const v8bf*)(pr);        // keys 0..7 | 16..23
            v8bf hi = *(const v8bf*)(pr + 8);    // keys 8..15 | 24..31
#pragma unroll
            for (int i = 0; i < 8; ++i) { pa[i] = lo[i]; pa[i + 8] = hi[i]; }
        }
        // ---- O += P * V  (Vt contiguous along keys) ------------------------
#pragma unroll
        for (int cg = 0; cg < 8; ++cg) {
            const bf16_t* vp = Vtb + (size_t)(cb + cg * 16 + lrow) * kN + kt + bsel;
            v16bf vb = *(const v16bf*)vp;
            o[cg] = WMMA_BF16(pa, vb, o[cg]);
        }
    }

    // ---- epilogue: out = gamma * (O / l) + x ------------------------------
    const float g = *gamma_p;
    float inv[8];
#pragma unroll
    for (int e = 0; e < 8; ++e) inv[e] = 1.0f / lsum[e];
#pragma unroll
    for (int cg = 0; cg < 8; ++cg) {
        const int c = cb + cg * 16 + lrow;
#pragma unroll
        for (int e = 0; e < 8; ++e) {
            const size_t idx = (size_t)(p0 + rb + e) * kC + c;
            out[idx] = g * (o[cg][e] * inv[e]) + x[idx];
        }
    }
}

// ---------------------------------------------------------------------------
// Host launcher
// ---------------------------------------------------------------------------
extern "C" void kernel_launch(void* const* d_in, const int* in_sizes, int n_in,
                              void* d_out, int out_size, void* d_ws, size_t ws_size,
                              hipStream_t stream) {
    const float* x     = (const float*)d_in[0];
    const float* Wb    = (const float*)d_in[1];
    const float* Wc    = (const float*)d_in[2];
    const float* Wd    = (const float*)d_in[3];
    const float* gamma = (const float*)d_in[4];
    float* out = (float*)d_out;

    char* ws = (char*)d_ws;
    // ws layout (bytes): Wt 640*512*2=640KB @0 | Qb 2MB @1MB | Kb 2MB @3MB | Vt 16MB @5MB
    bf16_t* Wt = (bf16_t*)(ws);
    bf16_t* Qb = (bf16_t*)(ws + (size_t)(1u << 20));
    bf16_t* Kb = (bf16_t*)(ws + (size_t)(3u << 20));
    bf16_t* Vt = (bf16_t*)(ws + (size_t)(5u << 20));

    hipLaunchKernelGGL(pam_prep_weights, dim3((kNCOL * kC + 255) / 256), dim3(256),
                       0, stream, Wb, Wc, Wd, Wt);
    // 1024 row-tiles x 10 col-tiles, 4 wave-tiles per block
    hipLaunchKernelGGL(pam_proj, dim3((1024 * 10) / 4), dim3(128),
                       0, stream, x, Wt, Qb, Kb, Vt);
    // one block per 16-query tile: B * N/16 = 1024
    hipLaunchKernelGGL(pam_attn, dim3(kB * (kN / 16)), dim3(128),
                       0, stream, x, gamma, Qb, Kb, Vt, out);
}
